// pointnet_2_55070070669895
// MI455X (gfx1250) — matrix operand order repaired and checked by reading, not compiled, not run on hardware
//
#include <hip/hip_runtime.h>

// ---------------- geometry constants (match reference) ----------------
#define B_CLOUDS 16
#define NPTS     4096
#define C_INF    64
#define MCENT    1024
#define KNBR     64
#define HIDF     64
#define OUTF     128
#define RAD2     0.04f      // 0.2^2
#define NBINS    512

typedef __attribute__((ext_vector_type(16))) _Float16 v16h;
typedef __attribute__((ext_vector_type(8)))  _Float16 v8h;
typedef __attribute__((ext_vector_type(8)))  float    v8f;

// =====================================================================
// Kernel 0: pre-swizzle a weight matrix [Ksrc x N] (f32 row-major) into
// WMMA B-fragment-linear f16 tiles. Tile (kt,nt) covers K=[kt*32,kt*32+32),
// N=[nt*16,nt*16+16). Fragment layout used here (16-bit B 32x16, contiguous k, same as load_a16):
//   lanes 0-15:  N=lane,    half j -> K = j
//   lanes 16-31: N=lane-16, half j -> K = 16+j
// Stored as Wf[tile*512 + lane*16 + j] so each lane loads a contiguous
// 32-byte v16h fragment.
// =====================================================================
__global__ void prep_w(const float* __restrict__ W, _Float16* __restrict__ Wf,
                       int Ksrc, int N, int ntiles) {
    int t    = blockIdx.x;          // tile index = kt*ntiles + nt
    int kt   = t / ntiles;
    int nt   = t % ntiles;
    int lane = threadIdx.x;         // 32 threads
    int n    = nt * 16 + (lane & 15);
    int kb   = kt * 32 + ((lane < 16) ? 0 : 16);
#pragma unroll
    for (int j = 0; j < 16; j++) {
        int K = kb + j;
        Wf[(size_t)t * 512 + lane * 16 + j] =
            (K < Ksrc) ? (_Float16)W[(size_t)K * N + n] : (_Float16)0.f;
    }
}

// =====================================================================
// Kernel 1: farthest point sampling. One block (256 threads) per cloud.
// Each thread owns 16 points: coords + running min-dist live in registers;
// LDS holds only the 256-wide argmax tree + chosen-point broadcast.
// Tie-break = smallest index (matches jnp.argmax first-occurrence).
// =====================================================================
__global__ __launch_bounds__(256) void fps_kernel(const float* __restrict__ point,
                                                  float* __restrict__ centers,
                                                  int* __restrict__ batch_out) {
    __shared__ float redv[256];
    __shared__ int   redi[256];
    __shared__ float cpt[3];

    int b = blockIdx.x, t = threadIdx.x;
    const float* P = point + (size_t)b * NPTS * 3;

    float lx[16], ly[16], lz[16], ld[16];
    int base = t * 16;
#pragma unroll
    for (int q = 0; q < 16; q++) {
        lx[q] = P[(base + q) * 3 + 0];
        ly[q] = P[(base + q) * 3 + 1];
        lz[q] = P[(base + q) * 3 + 2];
    }
    if (t == 0) {
        cpt[0] = lx[0]; cpt[1] = ly[0]; cpt[2] = lz[0];
        centers[(size_t)(b * MCENT) * 3 + 0] = lx[0];
        centers[(size_t)(b * MCENT) * 3 + 1] = ly[0];
        centers[(size_t)(b * MCENT) * 3 + 2] = lz[0];
        batch_out[b * MCENT] = b;
    }
    __syncthreads();
    float cx = cpt[0], cy = cpt[1], cz = cpt[2];
#pragma unroll
    for (int q = 0; q < 16; q++) {
        float dx = lx[q] - cx, dy = ly[q] - cy, dz = lz[q] - cz;
        ld[q] = dx * dx + dy * dy + dz * dz;
    }

    for (int m = 1; m < MCENT; m++) {
        float bv = -1.f; int bq = 0;
#pragma unroll
        for (int q = 0; q < 16; q++)
            if (ld[q] > bv) { bv = ld[q]; bq = q; }
        redv[t] = bv; redi[t] = base + bq;
        __syncthreads();
        for (int s = 128; s > 0; s >>= 1) {
            if (t < s) {
                float ov = redv[t + s]; int oi = redi[t + s];
                if (ov > redv[t] || (ov == redv[t] && oi < redi[t])) {
                    redv[t] = ov; redi[t] = oi;
                }
            }
            __syncthreads();
        }
        int nxt = redi[0];
        if ((nxt >> 4) == t) {
            int q = nxt & 15;
            cpt[0] = lx[q]; cpt[1] = ly[q]; cpt[2] = lz[q];
            centers[(size_t)(b * MCENT + m) * 3 + 0] = lx[q];
            centers[(size_t)(b * MCENT + m) * 3 + 1] = ly[q];
            centers[(size_t)(b * MCENT + m) * 3 + 2] = lz[q];
            batch_out[b * MCENT + m] = b;
        }
        __syncthreads();
        cx = cpt[0]; cy = cpt[1]; cz = cpt[2];
#pragma unroll
        for (int q = 0; q < 16; q++) {
            float dx = lx[q] - cx, dy = ly[q] - cy, dz = lz[q] - cz;
            float nd = dx * dx + dy * dy + dz * dz;
            ld[q] = fminf(ld[q], nd);
        }
    }
}

// =====================================================================
// Kernel 2: radius neighbors, K nearest in-ball via 512-bin distance
// histogram. Selected SET is deterministic (cut-bin resolved by index
// rank); slot order is irrelevant to the max-aggregation downstream.
// One block (128 threads) per center; each thread owns a contiguous
// chunk of 32 points so index-order ranks are well defined.
// =====================================================================
__global__ __launch_bounds__(128) void radius_kernel(const float* __restrict__ point,
                                                     const float* __restrict__ centers,
                                                     int* __restrict__ nbr_idx,
                                                     int* __restrict__ nbr_cnt) {
    __shared__ int sBins[NBINS];
    __shared__ int sInfo[3];     // cut bin, quota, total
    __shared__ int sBase[128];
    __shared__ int sPos;

    int c = blockIdx.x, b = c >> 10, t = threadIdx.x;
    float cx = centers[3 * c + 0], cy = centers[3 * c + 1], cz = centers[3 * c + 2];
    const float* P = point + (size_t)b * NPTS * 3;

    for (int i = t; i < NBINS; i += 128) sBins[i] = 0;
    if (t == 0) sPos = 0;
    __syncthreads();

    const float scale = (float)NBINS / RAD2;
    int i0 = t * 32;

    // pass 1: bin counts
    for (int q = 0; q < 32; q++) {
        int i = i0 + q;
        float dx = P[i * 3] - cx, dy = P[i * 3 + 1] - cy, dz = P[i * 3 + 2] - cz;
        float d2 = dx * dx + dy * dy + dz * dz;
        if (d2 < RAD2) {
            int bi = (int)(d2 * scale); if (bi > NBINS - 1) bi = NBINS - 1;
            atomicAdd(&sBins[bi], 1);
        }
    }
    __syncthreads();

    if (t == 0) {
        int acc = 0, cb = NBINS, qq = 0;
        for (int i = 0; i < NBINS; i++) {
            int v = sBins[i];
            if (cb == NBINS && acc + v > KNBR) { cb = i; qq = KNBR - acc; }
            acc += v;
        }
        sInfo[0] = cb; sInfo[1] = qq; sInfo[2] = acc;
        nbr_cnt[c] = (acc < KNBR) ? acc : KNBR;
    }
    __syncthreads();
    int cb = sInfo[0], quota = sInfo[1];

    // pass 2a: per-thread cut-bin counts (index order within chunk)
    int lc = 0;
    for (int q = 0; q < 32; q++) {
        int i = i0 + q;
        float dx = P[i * 3] - cx, dy = P[i * 3 + 1] - cy, dz = P[i * 3 + 2] - cz;
        float d2 = dx * dx + dy * dy + dz * dz;
        if (d2 < RAD2) {
            int bi = (int)(d2 * scale); if (bi > NBINS - 1) bi = NBINS - 1;
            if (bi == cb) lc++;
        }
    }
    sBase[t] = lc;
    __syncthreads();
    if (t == 0) {
        int acc = 0;
        for (int i = 0; i < 128; i++) { int v = sBase[i]; sBase[i] = acc; acc += v; }
    }
    __syncthreads();

    // pass 2b: emit selected indices
    int crank = sBase[t];
    for (int q = 0; q < 32; q++) {
        int i = i0 + q;
        float dx = P[i * 3] - cx, dy = P[i * 3 + 1] - cy, dz = P[i * 3 + 2] - cz;
        float d2 = dx * dx + dy * dy + dz * dz;
        if (d2 < RAD2) {
            int bi = (int)(d2 * scale); if (bi > NBINS - 1) bi = NBINS - 1;
            bool sel = false;
            if (bi < cb) sel = true;
            else if (bi == cb) { if (crank < quota) sel = true; crank++; }
            if (sel) {
                int pos = atomicAdd(&sPos, 1);
                nbr_idx[(size_t)c * KNBR + pos] = i;
            }
        }
    }
    __syncthreads();
    int total = sInfo[2];
    int cnt = (total < KNBR) ? total : KNBR;
    for (int k = t; k < KNBR; k += 128)
        if (k >= cnt) nbr_idx[(size_t)c * KNBR + k] = 0;
}

// =====================================================================
// A-fragment loader: 16-bit A 16x32 layout. Per lane, K data lives
// in two contiguous 8-half (16B) chunks: [kb..kb+7] and [kb+8..kb+15]
// with kb = k0 + 16*(lane>=16), row = row0 + (lane&15).
// =====================================================================
__device__ __forceinline__ v16h load_a16(const _Float16* base, int row0, int k0,
                                         int ld, int lane) {
    int r  = lane & 15;
    int hi = (lane >> 4) & 1;
    const _Float16* p = base + (row0 + r) * ld + k0 + hi * 16;
    v8h lo = *(const v8h*)p;
    v8h h2 = *(const v8h*)(p + 8);
    v16h o;
#pragma unroll
    for (int i = 0; i < 8; i++) { o[i] = lo[i]; o[8 + i] = h2[i]; }
    return o;
}

// =====================================================================
// Kernel 3: PointConv MLP + masked max aggregation. One block (4 waves)
// per center; wave w owns neighbor rows [16w,16w+16). 36 WMMAs per wave.
// =====================================================================
__global__ __launch_bounds__(128) void pointconv_kernel(
    const float* __restrict__ feat, const float* __restrict__ point,
    const float* __restrict__ centers,
    const int* __restrict__ nbr_idx, const int* __restrict__ nbr_cnt,
    const _Float16* __restrict__ W1f, const _Float16* __restrict__ W2f,
    const _Float16* __restrict__ W3f,
    const float* __restrict__ b1, const float* __restrict__ b2,
    const float* __restrict__ b3,
    float* __restrict__ out) {
    constexpr int LDA = 104;   // msg row stride (96 data + pad), 16B-aligned rows
    constexpr int LDH = 72;    // hidden row stride (64 data + pad)
    __shared__ __align__(32) _Float16 sMsg[64 * LDA];
    __shared__ __align__(32) _Float16 sH1[64 * LDH];
    __shared__ __align__(32) _Float16 sH2[64 * LDH];
    __shared__ float sMax[8 * OUTF];

    int c = blockIdx.x, b = c >> 10;
    int tid = threadIdx.x, lane = tid & 31, wv = tid >> 5;
    int cnt = nbr_cnt[c];
    float cx = centers[3 * c + 0], cy = centers[3 * c + 1], cz = centers[3 * c + 2];

    // ---- gather msg = [feat_j(64) | pos_j - center(3) | zeros(29)] (f16) ----
    {
        int row = tid >> 1, half = tid & 1;
        int j = nbr_idx[(size_t)c * KNBR + row];
        size_t gp = (size_t)b * NPTS + j;
        const float* fr = feat + gp * C_INF;
        _Float16* dst = sMsg + row * LDA;
        if (half == 0) {
#pragma unroll
            for (int q = 0; q < 32; q++) dst[q] = (_Float16)fr[q];
        } else {
#pragma unroll
            for (int q = 32; q < 64; q++) dst[q] = (_Float16)fr[q];
            const float* pr = point + gp * 3;
            dst[64] = (_Float16)(pr[0] - cx);
            dst[65] = (_Float16)(pr[1] - cy);
            dst[66] = (_Float16)(pr[2] - cz);
#pragma unroll
            for (int q = 67; q < 96; q++) dst[q] = (_Float16)0.f;
        }
    }
    __syncthreads();

    int r16 = lane & 15, hi = lane >> 4;
    int row0 = wv * 16;
    v8f zero = {0.f, 0.f, 0.f, 0.f, 0.f, 0.f, 0.f, 0.f};

    // ---- layer 1: [64 x 96] @ [96 x 64] ----
    {
        v8f acc[4];
#pragma unroll
        for (int nt = 0; nt < 4; nt++) acc[nt] = zero;
#pragma unroll
        for (int kt = 0; kt < 3; kt++) {
            v16h a = load_a16(sMsg, row0, kt * 32, LDA, lane);
#pragma unroll
            for (int nt = 0; nt < 4; nt++) {
                v16h bw = *(const v16h*)(W1f + ((kt * 4 + nt) * 512 + lane * 16));
                acc[nt] = __builtin_amdgcn_wmma_f32_16x16x32_f16(
                    false, a, false, bw, (short)0, acc[nt], false, false);
            }
        }
#pragma unroll
        for (int nt = 0; nt < 4; nt++) {
            int n = nt * 16 + r16;
            float bb = b1[n];
#pragma unroll
            for (int r = 0; r < 8; r++) {
                float v = acc[nt][r] + bb;
                v = v > 0.f ? v : 0.f;
                sH1[(row0 + hi * 8 + r) * LDH + n] = (_Float16)v;
            }
        }
    }
    __syncthreads();

    // ---- layer 2: [64 x 64] @ [64 x 64] ----
    {
        v8f acc[4];
#pragma unroll
        for (int nt = 0; nt < 4; nt++) acc[nt] = zero;
#pragma unroll
        for (int kt = 0; kt < 2; kt++) {
            v16h a = load_a16(sH1, row0, kt * 32, LDH, lane);
#pragma unroll
            for (int nt = 0; nt < 4; nt++) {
                v16h bw = *(const v16h*)(W2f + ((kt * 4 + nt) * 512 + lane * 16));
                acc[nt] = __builtin_amdgcn_wmma_f32_16x16x32_f16(
                    false, a, false, bw, (short)0, acc[nt], false, false);
            }
        }
#pragma unroll
        for (int nt = 0; nt < 4; nt++) {
            int n = nt * 16 + r16;
            float bb = b2[n];
#pragma unroll
            for (int r = 0; r < 8; r++) {
                float v = acc[nt][r] + bb;
                v = v > 0.f ? v : 0.f;
                sH2[(row0 + hi * 8 + r) * LDH + n] = (_Float16)v;
            }
        }
    }
    __syncthreads();

    // ---- layer 3: [64 x 64] @ [64 x 128], fused mask + row-max ----
    {
        v8f acc[8];
#pragma unroll
        for (int nt = 0; nt < 8; nt++) acc[nt] = zero;
#pragma unroll
        for (int kt = 0; kt < 2; kt++) {
            v16h a = load_a16(sH2, row0, kt * 32, LDH, lane);
#pragma unroll
            for (int nt = 0; nt < 8; nt++) {
                v16h bw = *(const v16h*)(W3f + ((kt * 8 + nt) * 512 + lane * 16));
                acc[nt] = __builtin_amdgcn_wmma_f32_16x16x32_f16(
                    false, a, false, bw, (short)0, acc[nt], false, false);
            }
        }
#pragma unroll
        for (int nt = 0; nt < 8; nt++) {
            int n = nt * 16 + r16;
            float bb = b3[n];
            float pm = -__builtin_inff();
#pragma unroll
            for (int r = 0; r < 8; r++) {
                int krow = row0 + hi * 8 + r;
                float v = acc[nt][r] + bb;
                v = v > 0.f ? v : 0.f;
                if (krow < cnt) pm = fmaxf(pm, v);
            }
            sMax[(wv * 2 + hi) * OUTF + n] = pm;
        }
    }
    __syncthreads();

    // ---- final max over the 8 partials per output channel ----
    {
        int n = tid;   // 128 threads == OUTF channels
        float m = sMax[n];
#pragma unroll
        for (int r = 1; r < 8; r++) m = fmaxf(m, sMax[r * OUTF + n]);
        out[(size_t)c * OUTF + n] = m;
    }
}

// =====================================================================
// host launch
// =====================================================================
extern "C" void kernel_launch(void* const* d_in, const int* in_sizes, int n_in,
                              void* d_out, int out_size, void* d_ws, size_t ws_size,
                              hipStream_t stream) {
    const float* xyz   = (const float*)d_in[0];   // [B*NP, 64] features
    const float* point = (const float*)d_in[1];   // [B*NP, 3]
    // d_in[2] = batch (implied by contiguous layout), d_in[3] = num_samples (==64)
    const float* W1 = (const float*)d_in[4];
    const float* b1 = (const float*)d_in[5];
    const float* W2 = (const float*)d_in[6];
    const float* b2 = (const float*)d_in[7];
    const float* W3 = (const float*)d_in[8];
    const float* b3 = (const float*)d_in[9];

    // workspace layout
    char* ws = (char*)d_ws;
    int* nbr_idx = (int*)ws;                                   // 16384*64*4 = 4 MB
    int* nbr_cnt = (int*)(ws + (size_t)4 * 1024 * 1024);       // 64 KB
    _Float16* W1f = (_Float16*)(ws + (size_t)4 * 1024 * 1024 + 64 * 1024); // 12 KB
    _Float16* W2f = (_Float16*)((char*)W1f + 12 * 512 * 2);    // 8 KB
    _Float16* W3f = (_Float16*)((char*)W2f + 8 * 512 * 2);     // 16 KB

    // d_out = [out(16384*128) | centers(16384*3) | batch(16384 as int bits)]
    float* out      = (float*)d_out;
    float* centers  = out + (size_t)B_CLOUDS * MCENT * OUTF;
    int*   batch_o  = (int*)(centers + (size_t)B_CLOUDS * MCENT * 3);

    // weight pre-swizzle into WMMA B-fragment order (f16)
    prep_w<<<dim3(12), dim3(32), 0, stream>>>(W1, W1f, 67, HIDF, 4);  // 3 kt x 4 nt
    prep_w<<<dim3(8),  dim3(32), 0, stream>>>(W2, W2f, 64, HIDF, 4);  // 2 kt x 4 nt
    prep_w<<<dim3(16), dim3(32), 0, stream>>>(W3, W3f, 64, OUTF, 8);  // 2 kt x 8 nt

    // FPS: centers + batch outputs
    fps_kernel<<<dim3(B_CLOUDS), dim3(256), 0, stream>>>(point, centers, batch_o);

    // radius neighbor selection
    radius_kernel<<<dim3(B_CLOUDS * MCENT), dim3(128), 0, stream>>>(
        point, centers, nbr_idx, nbr_cnt);

    // WMMA MLP + masked max aggregation
    pointconv_kernel<<<dim3(B_CLOUDS * MCENT), dim3(128), 0, stream>>>(
        xyz, point, centers, nbr_idx, nbr_cnt, W1f, W2f, W3f, b1, b2, b3, out);
}
